// Decoder_76613626626169
// MI455X (gfx1250) — matrix operand order repaired and checked by reading, not compiled, not run on hardware
//
#include <hip/hip_runtime.h>
#include <hip/hip_bf16.h>
#include <math.h>

// Problem dims
#define VDIM  32000
#define EDIM  512
#define H2    1024
#define SLEN  64
#define TLEN  64
#define BATCH 64
#define H4    4096
#define KX    2560   // xcat width: emb(512) | attn(1024) | h(1024)
#define KA    2048   // acat width: ws(1024) | h(1024)
#define KIH   1536   // E + H2 (W_ih inner dim)

// Toggle for the CDNA5 async-to-LDS staging path (inline asm).
#define TRY_ASYNC 1

typedef __attribute__((ext_vector_type(16))) __bf16 v16bf;
typedef __attribute__((ext_vector_type(8)))  float  v8f;

union ABf { uint4 q[2]; v16bf v; };

__device__ __forceinline__ unsigned short f2bf(float f) {
  unsigned int u = __float_as_uint(f);
  unsigned int r = (u + 0x7FFFu + ((u >> 16) & 1u)) >> 16;  // RNE
  return (unsigned short)r;
}

__device__ __forceinline__ float sigmoidf_(float x) { return 1.f / (1.f + expf(-x)); }

#define WMMA_BF16(a, b, c) \
  __builtin_amdgcn_wmma_f32_16x16x32_bf16(false, (a), false, (b), (short)0, (c), false, false)

// ---------------------------------------------------------------------------
// Multi-tile WMMA cores (one wave, 16x16 f32 tiles, fp32 accumulate).
// A lane striping (16-bit A 16x32): lane L holds row M=L&15; halves 0..15
// contiguous at K = k0+(L>>4)*16.
// B lane striping (32x16): lane L holds col N=L&15; halves 0..15 contiguous
// at K = k0+(L>>4)*16.  arow*/brow* are pre-offset per-lane pointers.
// ---------------------------------------------------------------------------
// MT output tiles sharing the SAME B fragment (M-blocking).
template <int MT>
__device__ __forceinline__ void wmma_shared_b(const unsigned short* const* arows,
                                              const unsigned short* brow,
                                              int K, v8f* acc) {
  for (int k0 = 0; k0 < K; k0 += 32) {
    ABf bb;
    const uint4* bp = (const uint4*)(brow + k0);
    bb.q[0] = bp[0];
    bb.q[1] = bp[1];
#pragma unroll
    for (int q = 0; q < MT; ++q) {
      ABf a;
      a.q[0] = *(const uint4*)(arows[q] + k0);
      a.q[1] = *(const uint4*)(arows[q] + k0 + 8);
      acc[q] = WMMA_BF16(a.v, bb.v, acc[q]);
    }
  }
}

// NT output tiles sharing the SAME A fragment (N-blocking).
template <int NT>
__device__ __forceinline__ void wmma_shared_a(const unsigned short* arow,
                                              const unsigned short* const* brows,
                                              int K, v8f* acc) {
  for (int k0 = 0; k0 < K; k0 += 32) {
    ABf a;
    a.q[0] = *(const uint4*)(arow + k0);
    a.q[1] = *(const uint4*)(arow + k0 + 8);
#pragma unroll
    for (int q = 0; q < NT; ++q) {
      ABf bb;
      const uint4* bp = (const uint4*)(brows[q] + k0);
      bb.q[0] = bp[0];
      bb.q[1] = bp[1];
      acc[q] = WMMA_BF16(a.v, bb.v, acc[q]);
    }
  }
}

// ---------------------------------------------------------------------------
// Setup kernels
// ---------------------------------------------------------------------------
__global__ void f32_to_bf16_kernel(const float* __restrict__ src,
                                   unsigned short* __restrict__ dst, int n) {
  int i = blockIdx.x * blockDim.x + threadIdx.x;
  if (i < n) dst[i] = f2bf(src[i]);
}

__global__ void build_wg_kernel(const float* __restrict__ W_ih, const float* __restrict__ W_hh,
                                unsigned short* __restrict__ wg) {
  int i = blockIdx.x * blockDim.x + threadIdx.x;
  if (i >= H4 * KX) return;
  int rr = i / KX, j = i % KX;
  float v = (j < KIH) ? W_ih[(size_t)rr * KIH + j]
                      : W_hh[(size_t)rr * H2 + (j - KIH)];
  wg[i] = f2bf(v);
}

__global__ void bias_comb_kernel(const float* __restrict__ b_ih, const float* __restrict__ b_hh,
                                 float* __restrict__ bg) {
  int i = blockIdx.x * blockDim.x + threadIdx.x;
  if (i < H4) bg[i] = b_ih[i] + b_hh[i];
}

__global__ void embed_kernel(const float* __restrict__ tab, const int* __restrict__ trg,
                             unsigned short* __restrict__ dst) {
  int i = blockIdx.x * blockDim.x + threadIdx.x;
  if (i >= TLEN * BATCH * EDIM) return;
  int e = i % EDIM;
  int tb = i / EDIM;                 // t*BATCH + b
  int tok = trg[tb];
  dst[i] = f2bf(tab[(size_t)tok * EDIM + e]);
}

__global__ void init_hc_kernel(const float* __restrict__ enc_h, const float* __restrict__ enc_c,
                               float* __restrict__ hbuf, float* __restrict__ cbuf,
                               unsigned short* __restrict__ h_bf) {
  int i = blockIdx.x * blockDim.x + threadIdx.x;
  if (i >= BATCH * H2) return;
  int b = i >> 10, j = i & (H2 - 1);
  const int Hr = H2 / 2;
  float hv = (j < Hr) ? enc_h[(size_t)b * Hr + j]
                      : enc_h[(size_t)BATCH * Hr + (size_t)b * Hr + (j - Hr)];
  float cv = (j < Hr) ? enc_c[(size_t)b * Hr + j]
                      : enc_c[(size_t)BATCH * Hr + (size_t)b * Hr + (j - Hr)];
  hbuf[i] = hv;
  cbuf[i] = cv;
  h_bf[i] = f2bf(hv);
}

// ctx_lin = seq_context(bf16) @ W_in^T : (4096 x 1024), 4 M-tiles per wave
__global__ void ctxlin_gemm_kernel(const unsigned short* __restrict__ sc_bf,
                                   const unsigned short* __restrict__ win_bf,
                                   float* __restrict__ ctxlin) {
  int gw = (blockIdx.x * blockDim.x + threadIdx.x) >> 5;   // 0..4095
  int lane = threadIdx.x & 31;
  int nt = gw & 63;         // 64 N tiles (H2/16)
  int mg = gw >> 6;         // 64 M groups of 4 tiles
  int g = lane >> 4, r = lane & 15;
  const unsigned short* arows[4];
#pragma unroll
  for (int q = 0; q < 4; ++q)
    arows[q] = sc_bf + (size_t)(mg * 64 + q * 16 + r) * H2 + g * 16;
  const unsigned short* brow = win_bf + (size_t)(nt * 16 + r) * H2 + g * 16;
  v8f acc[4] = {};
  wmma_shared_b<4>(arows, brow, H2, acc);
  int n = nt * 16 + r;
#pragma unroll
  for (int q = 0; q < 4; ++q)
    for (int rr = 0; rr < 8; ++rr)
      ctxlin[(size_t)(mg * 64 + q * 16 + rr + g * 8) * H2 + n] = acc[q][rr];
}

// ---------------------------------------------------------------------------
// Per-step kernels
// ---------------------------------------------------------------------------
// One block per batch row: attention scores, normalize, weighted sum; builds
// acat = bf16[ws|h] and fills xcat emb/h segments.
__global__ void attn_step_kernel(const float* __restrict__ ctxlin,
                                 const float* __restrict__ seq_context,
                                 const float* __restrict__ src_mask,
                                 const float* __restrict__ hbuf,
                                 const unsigned short* __restrict__ h_bf,
                                 const unsigned short* __restrict__ emb_bf_t,
                                 unsigned short* __restrict__ acat,
                                 unsigned short* __restrict__ xcat) {
  int b = blockIdx.x;
  int tid = threadIdx.x, lane = tid & 31, wid = tid >> 5;
  __shared__ float sh_h[H2];
  __shared__ float sh_w[SLEN];
  __shared__ float sh_denom;
#if TRY_ASYNC
  {
    // CDNA5 async copy: stage h[b] (4KB) global -> LDS, tracked by ASYNCcnt.
    unsigned loff = (unsigned)(unsigned long long)&sh_h[tid * 4];
    const float* gp = hbuf + (size_t)b * H2 + (size_t)tid * 4;
    asm volatile("global_load_async_to_lds_b128 %0, %1, off"
                 :: "v"(loff), "v"(gp) : "memory");
    asm volatile("s_wait_asynccnt 0x0" ::: "memory");
  }
  __syncthreads();
#else
  for (int i = tid; i < H2; i += 256) sh_h[i] = hbuf[(size_t)b * H2 + i];
  __syncthreads();
#endif
  // each of 8 waves handles 8 source positions
  for (int k = 0; k < 8; ++k) {
    int s = wid * 8 + k;
    const float* cl = ctxlin + ((size_t)s * BATCH + b) * H2;
    float p = 0.f;
    for (int i = lane; i < H2; i += 32) p += cl[i] * sh_h[i];
    for (int o = 16; o > 0; o >>= 1) p += __shfl_xor(p, o);
    if (lane == 0) sh_w[s] = expf(p) * src_mask[s * BATCH + b];
  }
  __syncthreads();
  if (tid == 0) {
    float d = 0.f;
    for (int s = 0; s < SLEN; ++s) d += sh_w[s];
    sh_denom = (d == 0.f) ? 1.f : d;
  }
  __syncthreads();
  float dn = sh_denom;
  if (tid < SLEN) sh_w[tid] = sh_w[tid] / dn;
  __syncthreads();
  for (int i = tid; i < H2; i += 256) {
    float acc = 0.f;
    for (int s = 0; s < SLEN; ++s)
      acc += sh_w[s] * seq_context[((size_t)s * BATCH + b) * H2 + i];
    acat[(size_t)b * KA + i]       = f2bf(acc);                  // ws
    acat[(size_t)b * KA + H2 + i]  = f2bf(sh_h[i]);              // h
    xcat[(size_t)b * KX + KIH + i] = h_bf[(size_t)b * H2 + i];   // h segment
  }
  for (int i = tid; i < EDIM; i += 256)
    xcat[(size_t)b * KX + i] = emb_bf_t[(size_t)b * EDIM + i];   // emb segment
}

// attn = tanh(acat @ W_attn^T) -> bf16 into xcat[:,512:1536]. 4 M-tiles/wave.
__global__ void attn_gemm_kernel(const unsigned short* __restrict__ acat,
                                 const unsigned short* __restrict__ wattn_bf,
                                 unsigned short* __restrict__ xcat) {
  int gw = (blockIdx.x * blockDim.x + threadIdx.x) >> 5;  // 0..63 (n-tile)
  int lane = threadIdx.x & 31;
  int g = lane >> 4, r = lane & 15;
  const unsigned short* arows[4];
#pragma unroll
  for (int q = 0; q < 4; ++q)
    arows[q] = acat + (size_t)(q * 16 + r) * KA + g * 16;
  const unsigned short* brow = wattn_bf + (size_t)(gw * 16 + r) * KA + g * 16;
  v8f acc[4] = {};
  wmma_shared_b<4>(arows, brow, KA, acc);
  int n = gw * 16 + r;
#pragma unroll
  for (int q = 0; q < 4; ++q)
    for (int rr = 0; rr < 8; ++rr) {
      int m = q * 16 + rr + g * 8;
      xcat[(size_t)m * KX + EDIM + n] = f2bf(tanhf(acc[q][rr]));
    }
}

// Fused gates GEMM + LSTM cell: one wave computes its i/f/g/o tiles for the
// same (batch-tile, j-tile) patch, then applies the cell update directly.
__global__ void gates_lstm_kernel(const unsigned short* __restrict__ xcat,
                                  const unsigned short* __restrict__ wg_bf,
                                  const float* __restrict__ bg,
                                  float* __restrict__ hbuf, float* __restrict__ cbuf,
                                  unsigned short* __restrict__ h_bf) {
  int gw = (blockIdx.x * blockDim.x + threadIdx.x) >> 5;  // 0..255
  int lane = threadIdx.x & 31;
  int mt = gw & 3;          // 4 batch tiles
  int jt = gw >> 2;         // 64 j tiles (H2/16)
  int g = lane >> 4, r = lane & 15;
  const unsigned short* arow = xcat + (size_t)(mt * 16 + r) * KX + g * 16;
  const unsigned short* brows[4];
#pragma unroll
  for (int q = 0; q < 4; ++q)   // gate q rows at q*H2 + j
    brows[q] = wg_bf + (size_t)(q * H2 + jt * 16 + r) * KX + g * 16;
  v8f acc[4] = {};
  wmma_shared_a<4>(arow, brows, KX, acc);
  int j = jt * 16 + r;
  float bi = bg[j], bf = bg[H2 + j], bgg = bg[2 * H2 + j], bo = bg[3 * H2 + j];
#pragma unroll
  for (int rr = 0; rr < 8; ++rr) {
    int m = mt * 16 + rr + g * 8;           // batch row
    size_t idx = (size_t)m * H2 + j;
    float ig = sigmoidf_(acc[0][rr] + bi);
    float fg = sigmoidf_(acc[1][rr] + bf);
    float gg = tanhf(acc[2][rr] + bgg);
    float og = sigmoidf_(acc[3][rr] + bo);
    float c = fg * cbuf[idx] + ig * gg;
    float h = og * tanhf(c);
    cbuf[idx] = c;
    hbuf[idx] = h;
    h_bf[idx] = f2bf(h);
  }
}

// Generator: logits = h_new @ W_gen^T + b_gen, 4 M-tiles per wave sharing B.
// Non-temporal stores keep the 8MB/step logit stream out of L2 so the
// bf16 weight set stays resident.
__global__ void gen_gemm_kernel(const unsigned short* __restrict__ h_bf,
                                const unsigned short* __restrict__ wgen_bf,
                                const float* __restrict__ b_gen,
                                float* __restrict__ outslab) {
  int gw = (blockIdx.x * blockDim.x + threadIdx.x) >> 5;  // 0..1999 (n-tile)
  int lane = threadIdx.x & 31;
  int g = lane >> 4, r = lane & 15;
  const unsigned short* arows[4];
#pragma unroll
  for (int q = 0; q < 4; ++q)
    arows[q] = h_bf + (size_t)(q * 16 + r) * H2 + g * 16;
  const unsigned short* brow = wgen_bf + (size_t)(gw * 16 + r) * H2 + g * 16;
  v8f acc[4] = {};
  wmma_shared_b<4>(arows, brow, H2, acc);
  int n = gw * 16 + r;
  float bn = b_gen[n];
#pragma unroll
  for (int q = 0; q < 4; ++q)
    for (int rr = 0; rr < 8; ++rr) {
      int m = q * 16 + rr + g * 8;
      __builtin_nontemporal_store(acc[q][rr] + bn, &outslab[(size_t)m * VDIM + n]);
    }
}

// In-place row-wise log-softmax over V; one block per (t,b) row; launched
// once over all 63*64 rows after the scan (rows are independent).
__global__ void logsoftmax_kernel(float* __restrict__ base) {
  float* row = base + (size_t)blockIdx.x * VDIM;
  int tid = threadIdx.x, lane = tid & 31, wid = tid >> 5;
  __shared__ float red[8];
  __shared__ float stat[2];
  float mx = -INFINITY;
  for (int i = tid; i < VDIM; i += 256) mx = fmaxf(mx, row[i]);
  for (int o = 16; o > 0; o >>= 1) mx = fmaxf(mx, __shfl_xor(mx, o));
  if (lane == 0) red[wid] = mx;
  __syncthreads();
  if (wid == 0) {
    float v = (lane < 8) ? red[lane] : -INFINITY;
    for (int o = 4; o > 0; o >>= 1) v = fmaxf(v, __shfl_xor(v, o));
    if (lane == 0) stat[0] = v;
  }
  __syncthreads();
  float m = stat[0];
  float s = 0.f;
  for (int i = tid; i < VDIM; i += 256) s += expf(row[i] - m);
  for (int o = 16; o > 0; o >>= 1) s += __shfl_xor(s, o);
  if (lane == 0) red[wid] = s;
  __syncthreads();
  if (wid == 0) {
    float v = (lane < 8) ? red[lane] : 0.f;
    for (int o = 4; o > 0; o >>= 1) v += __shfl_xor(v, o);
    if (lane == 0) stat[1] = logf(v);
  }
  __syncthreads();
  float ls = stat[1];
  for (int i = tid; i < VDIM; i += 256)
    __builtin_nontemporal_store(row[i] - m - ls, &row[i]);
}

// ---------------------------------------------------------------------------
extern "C" void kernel_launch(void* const* d_in, const int* in_sizes, int n_in,
                              void* d_out, int out_size, void* d_ws, size_t ws_size,
                              hipStream_t stream) {
  const float* seq_context = (const float*)d_in[0];
  const float* src_mask    = (const float*)d_in[1];
  const int*   seq_trg     = (const int*)d_in[2];
  const float* enc_h       = (const float*)d_in[3];
  const float* enc_c       = (const float*)d_in[4];
  const float* emb_table   = (const float*)d_in[5];
  const float* W_in        = (const float*)d_in[6];
  const float* W_attn      = (const float*)d_in[7];
  const float* W_ih        = (const float*)d_in[8];
  const float* W_hh        = (const float*)d_in[9];
  const float* b_ih        = (const float*)d_in[10];
  const float* b_hh        = (const float*)d_in[11];
  const float* W_gen       = (const float*)d_in[12];
  const float* b_gen       = (const float*)d_in[13];
  float* out = (float*)d_out;

  char* w = (char*)d_ws;
  size_t off = 0;
  auto take = [&](size_t bytes) -> char* {
    char* p = w + off;
    off = (off + bytes + 255) & ~(size_t)255;
    return p;
  };
  unsigned short* wgen_bf  = (unsigned short*)take((size_t)VDIM * H2 * 2);
  unsigned short* wattn_bf = (unsigned short*)take((size_t)H2 * KA * 2);
  unsigned short* wg_bf    = (unsigned short*)take((size_t)H4 * KX * 2);
  float*          biasg    = (float*)take((size_t)H4 * 4);
  unsigned short* emb_bf   = (unsigned short*)take((size_t)TLEN * BATCH * EDIM * 2);
  unsigned short* sc_bf    = (unsigned short*)take((size_t)SLEN * BATCH * H2 * 2);
  unsigned short* win_bf   = (unsigned short*)take((size_t)H2 * H2 * 2);
  float*          ctxlin   = (float*)take((size_t)SLEN * BATCH * H2 * 4);
  float*          hbuf     = (float*)take((size_t)BATCH * H2 * 4);
  float*          cbuf     = (float*)take((size_t)BATCH * H2 * 4);
  unsigned short* h_bf     = (unsigned short*)take((size_t)BATCH * H2 * 2);
  unsigned short* acat     = (unsigned short*)take((size_t)BATCH * KA * 2);
  unsigned short* xcat     = (unsigned short*)take((size_t)BATCH * KX * 2);

  #define GRID1(n) dim3((unsigned)(((n) + 255) / 256))

  // ---- setup: weight/activation conversion + precompute ----
  f32_to_bf16_kernel<<<GRID1(VDIM * H2), 256, 0, stream>>>(W_gen, wgen_bf, VDIM * H2);
  f32_to_bf16_kernel<<<GRID1(H2 * KA), 256, 0, stream>>>(W_attn, wattn_bf, H2 * KA);
  f32_to_bf16_kernel<<<GRID1(SLEN * BATCH * H2), 256, 0, stream>>>(seq_context, sc_bf,
                                                                   SLEN * BATCH * H2);
  f32_to_bf16_kernel<<<GRID1(H2 * H2), 256, 0, stream>>>(W_in, win_bf, H2 * H2);
  build_wg_kernel<<<GRID1(H4 * KX), 256, 0, stream>>>(W_ih, W_hh, wg_bf);
  bias_comb_kernel<<<GRID1(H4), 256, 0, stream>>>(b_ih, b_hh, biasg);
  embed_kernel<<<GRID1(TLEN * BATCH * EDIM), 256, 0, stream>>>(emb_table, seq_trg, emb_bf);
  init_hc_kernel<<<GRID1(BATCH * H2), 256, 0, stream>>>(enc_h, enc_c, hbuf, cbuf, h_bf);
  // ctx_lin: 64 m-groups x 64 n-tiles = 4096 waves -> 512 blocks
  ctxlin_gemm_kernel<<<512, 256, 0, stream>>>(sc_bf, win_bf, ctxlin);

  // ---- sequential decode: steps t = 1 .. T-1 (4 launches/step) ----
  for (int t = 1; t < TLEN; ++t) {
    attn_step_kernel<<<BATCH, 256, 0, stream>>>(
        ctxlin, seq_context, src_mask, hbuf, h_bf,
        emb_bf + (size_t)(t - 1) * BATCH * EDIM, acat, xcat);
    // 64 n-tile jobs (4 M-tiles each) = 64 waves -> 8 blocks
    attn_gemm_kernel<<<8, 256, 0, stream>>>(acat, wattn_bf, xcat);
    // 4 batch-tiles x 64 j-tiles = 256 waves -> 32 blocks (GEMM+LSTM fused)
    gates_lstm_kernel<<<32, 256, 0, stream>>>(xcat, wg_bf, biasg, hbuf, cbuf, h_bf);
    // 2000 n-tile jobs (4 M-tiles each) = 2000 waves -> 250 blocks
    gen_gemm_kernel<<<250, 256, 0, stream>>>(
        h_bf, wgen_bf, b_gen, out + (size_t)(t - 1) * BATCH * VDIM);
  }
  // One deferred launch for all (T-1)*B independent log-softmax rows.
  logsoftmax_kernel<<<(TLEN - 1) * BATCH, 256, 0, stream>>>(out);
  #undef GRID1
}
